// SupConLoss_16836271800363
// MI455X (gfx1250) — matrix operand order repaired and checked by reading, not compiled, not run on hardware
//
#include <hip/hip_runtime.h>

// SupConLoss fused, split-bf16 precision:
//   normalize rows -> split fp32 into (hi,lo) bf16 planes
//   sim = xn@xn^T via 3x V_WMMA_F32_16X16X32_BF16 per K=32 chunk (hi*hi + hi*lo + lo*hi)
//   staging via GLOBAL_LOAD_ASYNC_TO_LDS_B128 (ASYNCcnt path), fallback to reg staging
//   fused epilogue: divisor[i] += exp(sim - diag), S1 += pos*sim, M += pos  (symmetric tiles x2)
//   finalize: loss = -(S1 - N*sum_i log divisor_i)/M

typedef __attribute__((ext_vector_type(4)))  float  v4f;
typedef __attribute__((ext_vector_type(8)))  float  v8f;
typedef __attribute__((ext_vector_type(4)))  int    v4i;
typedef __attribute__((ext_vector_type(8)))  __bf16 v8bf;
typedef __attribute__((ext_vector_type(16))) __bf16 v16bf;

#define TM 128
#define TN 128
#define KC 32    // bf16 K per LDS chunk = one WMMA K
#define LDB 40   // LDS row stride in bf16 units (80B: bank conflict-free, 16B aligned)

#if __has_builtin(__builtin_amdgcn_global_load_async_to_lds_b128) && \
    __has_builtin(__builtin_amdgcn_s_wait_asynccnt)
#define HAVE_ASYNC_LDS 1
typedef __attribute__((address_space(1))) v4i gv4i;   // global (AS1) 16B chunk
typedef __attribute__((address_space(3))) v4i lv4i;   // LDS    (AS3) 16B chunk
#else
#define HAVE_ASYNC_LDS 0
#endif

__device__ __forceinline__ float wave_sum(float v) {
#pragma unroll
    for (int o = 16; o > 0; o >>= 1) v += __shfl_xor(v, o);
    return v;
}

__device__ __forceinline__ unsigned short f2bf(float f) {  // round-to-nearest-even
    unsigned u = __float_as_uint(f);
    u += 0x7FFFu + ((u >> 16) & 1u);
    return (unsigned short)(u >> 16);
}
__device__ __forceinline__ float bf2f(unsigned short h) {
    return __uint_as_float((unsigned)h << 16);
}

__global__ __launch_bounds__(256) void init_ws(float* ws, int n) {
    int i = blockIdx.x * blockDim.x + threadIdx.x;
    if (i < n) ws[i] = 0.0f;
}

// One wave per row: rnorm, then write hi/lo bf16 planes of the normalized row.
__global__ __launch_bounds__(256) void normalize_split(
        const float* __restrict__ x, unsigned short* __restrict__ xh,
        unsigned short* __restrict__ xl, int N, int D) {
    int row  = blockIdx.x * 8 + (threadIdx.x >> 5);
    int lane = threadIdx.x & 31;
    if (row >= N) return;
    const float* xr = x + (size_t)row * D;
    float ss = 0.f;
    for (int c = lane * 4; c < D; c += 128) {
        v4f v = *(const v4f*)(xr + c);
        ss += v.x * v.x + v.y * v.y + v.z * v.z + v.w * v.w;
    }
    ss = wave_sum(ss);
    float rn = 1.0f / fmaxf(sqrtf(ss), 1e-8f);
    for (int c = lane * 4; c < D; c += 128) {
        v4f v = *(const v4f*)(xr + c);
        float vv[4] = {v.x * rn, v.y * rn, v.z * rn, v.w * rn};
        unsigned short h[4], l[4];
#pragma unroll
        for (int t = 0; t < 4; ++t) {
            h[t] = f2bf(vv[t]);
            l[t] = f2bf(vv[t] - bf2f(h[t]));
        }
        uint2 ph = {(unsigned)h[0] | ((unsigned)h[1] << 16),
                    (unsigned)h[2] | ((unsigned)h[3] << 16)};
        uint2 pl = {(unsigned)l[0] | ((unsigned)l[1] << 16),
                    (unsigned)l[2] | ((unsigned)l[3] << 16)};
        *(uint2*)(xh + (size_t)row * D + c) = ph;
        *(uint2*)(xl + (size_t)row * D + c) = pl;
    }
}

// Upper-triangle 128x128 tiles of sim via split-bf16 WMMA; fused loss epilogue.
__global__ __launch_bounds__(256) void supcon_tile(
        const unsigned short* __restrict__ xh, const unsigned short* __restrict__ xl,
        const int* __restrict__ labels,
        float* __restrict__ divisor, float* __restrict__ s1cnt,
        int N, int D) {
    const int bi = blockIdx.y;
    const int bj = blockIdx.x;
    if (bj < bi) return;                  // symmetric: upper triangle only
    const bool diag = (bi == bj);

    __shared__ __align__(16) unsigned short AhS[TM * LDB];
    __shared__ __align__(16) unsigned short AlS[TM * LDB];
    __shared__ __align__(16) unsigned short BhS[TN * LDB];
    __shared__ __align__(16) unsigned short BlS[TN * LDB];

    const int tid  = threadIdx.x;
    const int lane = tid & 31;
    const int wave = tid >> 5;
    const int half = lane >> 4;           // WMMA lane-half
    const int l16  = lane & 15;
    const int wm0  = (wave & 1) * 64;     // 2 waves along M
    const int wn0  = (wave >> 1) * 32;    // 4 waves along N
    const int rowA0 = bi * TM;
    const int rowB0 = bj * TN;

    v8f c[4][2];
#pragma unroll
    for (int mi = 0; mi < 4; ++mi)
#pragma unroll
        for (int ni = 0; ni < 2; ++ni)
            c[mi][ni] = v8f{};

    for (int kc = 0; kc < D; kc += KC) {
        __syncthreads();
        // Stage 4 planes: 128 rows x 32 bf16 (64B) each = 512 16B-chunks per plane.
#pragma unroll
        for (int v = 0; v < 2; ++v) {
            int idx = tid + 256 * v;
            int r = idx >> 2;             // 0..127
            int q = idx & 3;              // 16B chunk in row
            size_t gA = (size_t)(rowA0 + r) * D + kc + q * 8;
            size_t gB = (size_t)(rowB0 + r) * D + kc + q * 8;
            unsigned lo = r * LDB + q * 8;  // LDS offset (ushort units)
#if HAVE_ASYNC_LDS
            // Direct global -> LDS DMA, no VGPR round-trip, tracked by ASYNCcnt.
            __builtin_amdgcn_global_load_async_to_lds_b128(
                (gv4i*)(xh + gA), (lv4i*)&AhS[lo], 0, 0);
            __builtin_amdgcn_global_load_async_to_lds_b128(
                (gv4i*)(xl + gA), (lv4i*)&AlS[lo], 0, 0);
            __builtin_amdgcn_global_load_async_to_lds_b128(
                (gv4i*)(xh + gB), (lv4i*)&BhS[lo], 0, 0);
            __builtin_amdgcn_global_load_async_to_lds_b128(
                (gv4i*)(xl + gB), (lv4i*)&BlS[lo], 0, 0);
#else
            *(uint4*)(&AhS[lo]) = *(const uint4*)(xh + gA);
            *(uint4*)(&AlS[lo]) = *(const uint4*)(xl + gA);
            *(uint4*)(&BhS[lo]) = *(const uint4*)(xh + gB);
            *(uint4*)(&BlS[lo]) = *(const uint4*)(xl + gB);
#endif
        }
#if HAVE_ASYNC_LDS
        __builtin_amdgcn_s_wait_asynccnt(0);   // this wave's DMAs have landed in LDS
#endif
        __syncthreads();                        // all waves' tiles visible

        // B fragments (32x16 bf16): lane=column l16, K = 16*half + 0..15 (32B contiguous).
        v16bf bh[2], bl[2];
#pragma unroll
        for (int ni = 0; ni < 2; ++ni) {
            const unsigned short* pb = &BhS[(wn0 + ni * 16 + l16) * LDB + 16 * half];
            v8bf y0 = *(const v8bf*)(pb);
            v8bf y1 = *(const v8bf*)(pb + 8);
            bh[ni] = __builtin_shufflevector(y0, y1, 0,1,2,3,4,5,6,7,8,9,10,11,12,13,14,15);
            const unsigned short* qb = &BlS[(wn0 + ni * 16 + l16) * LDB + 16 * half];
            v8bf z0 = *(const v8bf*)(qb);
            v8bf z1 = *(const v8bf*)(qb + 8);
            bl[ni] = __builtin_shufflevector(z0, z1, 0,1,2,3,4,5,6,7,8,9,10,11,12,13,14,15);
        }

        // A fragments (16x32 bf16): lane=row l16, K = 16*half + 0..15 (32B contiguous).
#pragma unroll
        for (int mi = 0; mi < 4; ++mi) {
            const unsigned short* pa = &AhS[(wm0 + mi * 16 + l16) * LDB];
            v8bf x0 = *(const v8bf*)(pa + 16 * half);
            v8bf x1 = *(const v8bf*)(pa + 16 * half + 8);
            v16bf ah = __builtin_shufflevector(x0, x1, 0,1,2,3,4,5,6,7,8,9,10,11,12,13,14,15);
            const unsigned short* qa = &AlS[(wm0 + mi * 16 + l16) * LDB];
            v8bf w0 = *(const v8bf*)(qa + 16 * half);
            v8bf w1 = *(const v8bf*)(qa + 16 * half + 8);
            v16bf al = __builtin_shufflevector(w0, w1, 0,1,2,3,4,5,6,7,8,9,10,11,12,13,14,15);
#pragma unroll
            for (int ni = 0; ni < 2; ++ni) {
                // x*y ~= hi*hi + hi*lo + lo*hi (fp32 accumulate)
                c[mi][ni] = __builtin_amdgcn_wmma_f32_16x16x32_bf16(
                    false, ah, false, bh[ni], (short)0, c[mi][ni], false, false);
                c[mi][ni] = __builtin_amdgcn_wmma_f32_16x16x32_bf16(
                    false, ah, false, bl[ni], (short)0, c[mi][ni], false, false);
                c[mi][ni] = __builtin_amdgcn_wmma_f32_16x16x32_bf16(
                    false, al, false, bh[ni], (short)0, c[mi][ni], false, false);
            }
        }
    }

    // Fused epilogue. C layout: VGPR r, lane L -> (M = r + 8*(L/16), N = L%16).
    float s1 = 0.f, cnt = 0.f;
    const float w = diag ? 1.0f : 2.0f;
#pragma unroll
    for (int mi = 0; mi < 4; ++mi) {
#pragma unroll
        for (int ni = 0; ni < 2; ++ni) {
            const int i0 = rowA0 + wm0 + mi * 16;
            const int jj = rowB0 + wn0 + ni * 16 + l16;   // this lane's column
            const int lj = labels[jj];
            float colExp = 0.f;
#pragma unroll
            for (int r = 0; r < 8; ++r) {
                const int   ii   = i0 + r + 8 * half;
                const float s    = c[mi][ni][r];
                const bool  self = (ii == jj);
                const bool  pos  = (labels[ii] == lj) && !self;
                if (pos) { s1 += w * s; cnt += w; }
                const float e = __expf(s - (self ? 1.0f : 0.0f));
                colExp += e;
                float re = e;                 // row-sum over the 16 lanes of this half
#pragma unroll
                for (int o = 1; o < 16; o <<= 1) re += __shfl_xor(re, o);
                if (l16 == 0) atomicAdd(&divisor[ii], re);
            }
            if (!diag) {                      // mirrored (j,i) elements -> divisor[j]
                colExp += __shfl_xor(colExp, 16);
                if (half == 0) atomicAdd(&divisor[jj], colExp);
            }
        }
    }
    s1  = wave_sum(s1);
    cnt = wave_sum(cnt);
    if (lane == 0) {
        atomicAdd(&s1cnt[0], s1);
        atomicAdd(&s1cnt[1], cnt);
    }
}

__global__ __launch_bounds__(256) void finalize(
        const float* __restrict__ divisor, const float* __restrict__ s1cnt,
        float* __restrict__ out, int N) {
    __shared__ float red[8];
    float ls = 0.f;
    for (int i = threadIdx.x; i < N; i += 256) ls += __logf(divisor[i]);
    ls = wave_sum(ls);
    const int wave = threadIdx.x >> 5, lane = threadIdx.x & 31;
    if (lane == 0) red[wave] = ls;
    __syncthreads();
    if (threadIdx.x == 0) {
        float s2 = 0.f;
#pragma unroll
        for (int i = 0; i < 8; ++i) s2 += red[i];
        out[0] = -(s1cnt[0] - (float)N * s2) / s1cnt[1];
    }
}

extern "C" void kernel_launch(void* const* d_in, const int* in_sizes, int n_in,
                              void* d_out, int out_size, void* d_ws, size_t ws_size,
                              hipStream_t stream) {
    const float* x      = (const float*)d_in[0];
    const int*   labels = (const int*)d_in[1];
    const int N = in_sizes[1];
    const int D = in_sizes[0] / N;

    // ws layout: [0,N) divisor f32 | [N,N+2) S1,M | +pad | xh bf16 N*D | xl bf16 N*D
    float* wsf     = (float*)d_ws;
    float* divisor = wsf;
    float* s1cnt   = wsf + N;
    unsigned short* xh = (unsigned short*)(wsf + N + 64);
    unsigned short* xl = xh + (size_t)N * D;

    init_ws<<<(N + 2 + 255) / 256, 256, 0, stream>>>(wsf, N + 2);
    normalize_split<<<(N + 7) / 8, 256, 0, stream>>>(x, xh, xl, N, D);
    dim3 grid(N / TN, N / TM);
    supcon_tile<<<grid, 256, 0, stream>>>(xh, xl, labels, divisor, s1cnt, N, D);
    finalize<<<1, 256, 0, stream>>>(divisor, s1cnt, (float*)d_out, N);
}
